// SPINN1_87703232184947
// MI455X (gfx1250) — matrix operand order repaired and checked by reading, not compiled, not run on hardware
//
#include <hip/hip_runtime.h>
#include <hip/hip_bf16.h>

typedef __attribute__((ext_vector_type(16))) __bf16 v16bf;
typedef __attribute__((ext_vector_type(8)))  __bf16 v8bf;
typedef __attribute__((ext_vector_type(8)))  float  v8f;

#define BB   128
#define TT   40
#define DH   512
#define TRK  256
#define CAP  (TT + 2)
#define TWO_D (2 * DH)          // 1024
#define KTRK (3 * DH + TRK)     // 1792
#define NTRK (4 * TRK)          // 1024
#define KRED (2 * DH + TRK)     // 1280
#define NRED (5 * DH)           // 2560

__device__ __forceinline__ __bf16 f2bf(float f) {
  union { float f; unsigned u; } x; x.f = f;
  unsigned r = (x.u + 0x7FFFu + ((x.u >> 16) & 1u)) >> 16;
  unsigned short s = (unsigned short)r;
  __bf16 b;
  __builtin_memcpy(&b, &s, 2);
  return b;
}

__device__ __forceinline__ float sigf(float x) {
  return 1.0f / (1.0f + __expf(-x));
}

// ---- WMMA fragment helpers (contiguous k order in A and B, wave32) ----

// A 16x32 bf16 from row-major [rows x lda]: lane holds row mb+(lane&15),
// K = 0..15 + 16*(lane>>4)  => two contiguous 16B loads.
__device__ __forceinline__ v16bf load_a_frag(const __bf16* A, int lda, int mb,
                                             int kb, int lane) {
  int m  = mb + (lane & 15);
  int kh = (lane >> 4) << 4;               // 0 or 16
  const __bf16* row = A + (long)m * lda + kb + kh;
  v8bf lo = *(const v8bf*)(row);           // K = kb+kh+0..7
  v8bf hi = *(const v8bf*)(row + 8);       // K = kb+kh+8..15
  return __builtin_shufflevector(lo, hi, 0,1,2,3,4,5,6,7,8,9,10,11,12,13,14,15);
}

// B 32x16 bf16 from pre-swizzled fragment-major storage:
// tile (kt/32, nb/16): lane = (n&15)|((k>>4&1)<<4), 16 contiguous K-elems.
__device__ __forceinline__ v16bf load_b_frag(const __bf16* W, int Ntiles,
                                             int kt, int nb, int lane) {
  long tile = (long)(kt >> 5) * Ntiles + (nb >> 4);
  const __bf16* p = W + (tile * 32 + lane) * 16;
  return *(const v16bf*)p;
}

__device__ __forceinline__ long btile_idx(int k, int n, int Ntiles) {
  long tile = (long)(k >> 5) * Ntiles + (n >> 4);
  int lane = (n & 15) | (((k >> 4) & 1) << 4);
  int elem = k & 15;
  return (tile * 32 + lane) * 16 + elem;
}

// ---- weight prep: f32 -> bf16, swizzled to B-fragment layout ----
__global__ __launch_bounds__(256) void spinn_prep(
    const float* __restrict__ W_ih, const float* __restrict__ W_hh,
    const float* __restrict__ W_left, const float* __restrict__ W_right,
    const float* __restrict__ W_track,
    __bf16* __restrict__ Wtrk, __bf16* __restrict__ Wred) {
  long idx = (long)blockIdx.x * 256 + threadIdx.x;
  const long n1 = (long)KTRK * NTRK;
  const long n2 = (long)KRED * NRED;
  if (idx < n1) {
    int k = (int)(idx / NTRK), n = (int)(idx % NTRK);
    float v = (k < 3 * DH) ? W_ih[(long)n * (3 * DH) + k]
                           : W_hh[(long)n * TRK + (k - 3 * DH)];
    Wtrk[btile_idx(k, n, NTRK / 16)] = f2bf(v);
  } else if (idx < n1 + n2) {
    long j = idx - n1;
    int k = (int)(j / NRED), n = (int)(j % NRED);
    float v;
    if (k < DH)            v = W_left [(long)n * DH + k];
    else if (k < 2 * DH)   v = W_right[(long)n * DH + (k - DH)];
    else                   v = W_track[(long)n * TRK + (k - 2 * DH)];
    Wred[btile_idx(k, n, NRED / 16)] = f2bf(v);
  }
}

// ---- state init ----
__global__ __launch_bounds__(256) void spinn_init(
    const float* __restrict__ buffers, float* __restrict__ stack,
    int* __restrict__ sptr, int* __restrict__ blen,
    float* __restrict__ th, float* __restrict__ tc) {
  int b = blockIdx.x;
  for (int i = threadIdx.x; i < TWO_D; i += 256) {
    float v = buffers[(long)b * TT * TWO_D + i];
    stack[((long)b * CAP + 0) * TWO_D + i] = v;
    stack[((long)b * CAP + 1) * TWO_D + i] = v;
  }
  for (int i = threadIdx.x; i < TRK; i += 256) {
    th[b * TRK + i] = 0.f;
    tc[b * TRK + i] = 0.f;
  }
  if (threadIdx.x == 0) { sptr[b] = 2; blen[b] = TT; }
}

// ---- per-step activation staging (bf16) ----
__global__ __launch_bounds__(256) void spinn_gather(
    const float* __restrict__ buffers, const float* __restrict__ stack,
    const int* __restrict__ sptr, const int* __restrict__ blen,
    const float* __restrict__ th,
    __bf16* __restrict__ xh, __bf16* __restrict__ ybuf) {
  int b  = blockIdx.x;
  int sp = sptr[b], bl = blen[b];
  const float* s1 = stack + ((long)b * CAP + (sp - 1)) * TWO_D;
  const float* s2 = stack + ((long)b * CAP + (sp - 2)) * TWO_D;
  const float* bt = buffers + ((long)b * TT + (bl - 1)) * TWO_D;
  for (int i = threadIdx.x; i < KTRK; i += 256) {
    float v;
    if (i < DH)           v = bt[i];
    else if (i < 2 * DH)  v = s1[i - DH];
    else if (i < 3 * DH)  v = s2[i - 2 * DH];
    else                  v = th[b * TRK + (i - 3 * DH)];
    xh[(long)b * KTRK + i] = f2bf(v);
  }
  for (int i = threadIdx.x; i < TWO_D; i += 256) {
    float v = (i < DH) ? s2[i] : s1[i - DH];
    ybuf[(long)b * TWO_D + i] = f2bf(v);
  }
}

// ---- tracker GEMM + LSTM cell (WMMA bf16, ping-pong pipelined K loop) ----
__global__ __launch_bounds__(256) void spinn_tracker(
    const __bf16* __restrict__ xh, const __bf16* __restrict__ Wtrk,
    const float* __restrict__ b_ih, const float* __restrict__ b_hh,
    const float* __restrict__ tc,
    float* __restrict__ th_new, float* __restrict__ tc_new) {
  int jt   = blockIdx.x;             // 0..15 : 16-col slice of the 256 tracker cols
  int wave = threadIdx.x >> 5;       // 0..7  : M-tile
  int lane = threadIdx.x & 31;
  int mb   = wave * 16;
  v8f acc[4] = {};

  v16bf af0, af1, bf0[4], bf1[4];
  af0 = load_a_frag(xh, KTRK, mb, 0, lane);
#pragma unroll
  for (int g = 0; g < 4; ++g)
    bf0[g] = load_b_frag(Wtrk, NTRK / 16, 0, g * TRK + jt * 16, lane);

  int kt = 0;
  // 56 K-steps total; body retires 2 per iteration with no register rotation
  for (; kt + 64 < KTRK; kt += 64) {
    af1 = load_a_frag(xh, KTRK, mb, kt + 32, lane);
#pragma unroll
    for (int g = 0; g < 4; ++g)
      bf1[g] = load_b_frag(Wtrk, NTRK / 16, kt + 32, g * TRK + jt * 16, lane);
#pragma unroll
    for (int g = 0; g < 4; ++g)
      acc[g] = __builtin_amdgcn_wmma_f32_16x16x32_bf16(
          false, af0, false, bf0[g], (short)0, acc[g], false, false);
    af0 = load_a_frag(xh, KTRK, mb, kt + 64, lane);
#pragma unroll
    for (int g = 0; g < 4; ++g)
      bf0[g] = load_b_frag(Wtrk, NTRK / 16, kt + 64, g * TRK + jt * 16, lane);
#pragma unroll
    for (int g = 0; g < 4; ++g)
      acc[g] = __builtin_amdgcn_wmma_f32_16x16x32_bf16(
          false, af1, false, bf1[g], (short)0, acc[g], false, false);
  }
  // epilogue: kt == KTRK - 64
  af1 = load_a_frag(xh, KTRK, mb, kt + 32, lane);
#pragma unroll
  for (int g = 0; g < 4; ++g)
    bf1[g] = load_b_frag(Wtrk, NTRK / 16, kt + 32, g * TRK + jt * 16, lane);
#pragma unroll
  for (int g = 0; g < 4; ++g)
    acc[g] = __builtin_amdgcn_wmma_f32_16x16x32_bf16(
        false, af0, false, bf0[g], (short)0, acc[g], false, false);
#pragma unroll
  for (int g = 0; g < 4; ++g)
    acc[g] = __builtin_amdgcn_wmma_f32_16x16x32_bf16(
        false, af1, false, bf1[g], (short)0, acc[g], false, false);

  int col = jt * 16 + (lane & 15);
  int rowAdd = (lane >> 4) * 8;
#pragma unroll
  for (int r = 0; r < 8; ++r) {
    int b = mb + r + rowAdd;
    float gi = acc[0][r] + b_ih[col]           + b_hh[col];
    float gf = acc[1][r] + b_ih[TRK + col]     + b_hh[TRK + col];
    float gg = acc[2][r] + b_ih[2 * TRK + col] + b_hh[2 * TRK + col];
    float go = acc[3][r] + b_ih[3 * TRK + col] + b_hh[3 * TRK + col];
    float tcv = sigf(gf) * tc[b * TRK + col] + sigf(gi) * tanhf(gg);
    float thv = sigf(go) * tanhf(tcv);
    tc_new[b * TRK + col] = tcv;
    th_new[b * TRK + col] = thv;
  }
}

// ---- logits, argmax, flags, state commit ----
__global__ __launch_bounds__(128) void spinn_logits(
    const float* __restrict__ th_new, const float* __restrict__ tc_new,
    const float* __restrict__ W_trans, const float* __restrict__ b_trans,
    const int* __restrict__ sptr, const int* __restrict__ blen,
    float* __restrict__ th, float* __restrict__ tc,
    __bf16* __restrict__ thb, float* __restrict__ out_step,
    int* __restrict__ shiftF, int* __restrict__ redF) {
  int b = threadIdx.x;
  float best = -3.4e38f; int bi = 0;
#pragma unroll
  for (int j = 0; j < 4; ++j) {
    float s = b_trans[j];
    for (int k = 0; k < TRK; ++k)
      s += th_new[b * TRK + k] * W_trans[j * TRK + k];
    out_step[b * 4 + j] = s;
    if (s > best) { best = s; bi = j; }   // first-max tie-break like jnp.argmax
  }
  shiftF[b] = (bi == 3 && blen[b] > 2) ? 1 : 0;
  redF[b]   = (bi == 2 && sptr[b] > 3) ? 1 : 0;
  for (int k = 0; k < TRK; ++k) {
    float v = th_new[b * TRK + k];
    th[b * TRK + k]  = v;
    tc[b * TRK + k]  = tc_new[b * TRK + k];
    thb[b * TRK + k] = f2bf(v);
  }
}

// ---- reduce (TreeLSTM) GEMM + conditional stack write (WMMA, ping-pong) ----
__global__ __launch_bounds__(256) void spinn_reduce(
    const __bf16* __restrict__ ybuf, const __bf16* __restrict__ thb,
    const __bf16* __restrict__ Wred, const float* __restrict__ b_left,
    const int* __restrict__ sptr, const int* __restrict__ redF,
    float* __restrict__ stack) {
  int jt   = blockIdx.x;             // 0..31 : 16-col slice of the 512 node cols
  int wave = threadIdx.x >> 5;
  int lane = threadIdx.x & 31;
  int mb   = wave * 16;
  v8f acc[5] = {};

  v16bf af0, af1, bf0[5], bf1[5];
  af0 = load_a_frag(ybuf, TWO_D, mb, 0, lane);
#pragma unroll
  for (int g = 0; g < 5; ++g)
    bf0[g] = load_b_frag(Wred, NRED / 16, 0, g * DH + jt * 16, lane);

  int kt = 0;
  // 40 K-steps total
  for (; kt + 64 < KRED; kt += 64) {
    int ka = kt + 32;
    af1 = (ka < 2 * DH) ? load_a_frag(ybuf, TWO_D, mb, ka, lane)
                        : load_a_frag(thb, TRK, mb, ka - 2 * DH, lane);
#pragma unroll
    for (int g = 0; g < 5; ++g)
      bf1[g] = load_b_frag(Wred, NRED / 16, ka, g * DH + jt * 16, lane);
#pragma unroll
    for (int g = 0; g < 5; ++g)
      acc[g] = __builtin_amdgcn_wmma_f32_16x16x32_bf16(
          false, af0, false, bf0[g], (short)0, acc[g], false, false);
    int kb = kt + 64;
    af0 = (kb < 2 * DH) ? load_a_frag(ybuf, TWO_D, mb, kb, lane)
                        : load_a_frag(thb, TRK, mb, kb - 2 * DH, lane);
#pragma unroll
    for (int g = 0; g < 5; ++g)
      bf0[g] = load_b_frag(Wred, NRED / 16, kb, g * DH + jt * 16, lane);
#pragma unroll
    for (int g = 0; g < 5; ++g)
      acc[g] = __builtin_amdgcn_wmma_f32_16x16x32_bf16(
          false, af1, false, bf1[g], (short)0, acc[g], false, false);
  }
  // epilogue: kt == KRED - 64
  af1 = load_a_frag(thb, TRK, mb, kt + 32 - 2 * DH, lane);
#pragma unroll
  for (int g = 0; g < 5; ++g)
    bf1[g] = load_b_frag(Wred, NRED / 16, kt + 32, g * DH + jt * 16, lane);
#pragma unroll
  for (int g = 0; g < 5; ++g)
    acc[g] = __builtin_amdgcn_wmma_f32_16x16x32_bf16(
        false, af0, false, bf0[g], (short)0, acc[g], false, false);
#pragma unroll
  for (int g = 0; g < 5; ++g)
    acc[g] = __builtin_amdgcn_wmma_f32_16x16x32_bf16(
        false, af1, false, bf1[g], (short)0, acc[g], false, false);

  int col = jt * 16 + (lane & 15);
  int rowAdd = (lane >> 4) * 8;
#pragma unroll
  for (int r = 0; r < 8; ++r) {
    int b  = mb + r + rowAdd;
    int sp = sptr[b];
    float*       s2 = stack + ((long)b * CAP + (sp - 2)) * TWO_D;
    const float* s1 = stack + ((long)b * CAP + (sp - 1)) * TWO_D;
    float ga = acc[0][r] + b_left[col];
    float gi = acc[1][r] + b_left[DH + col];
    float f1 = acc[2][r] + b_left[2 * DH + col];
    float f2 = acc[3][r] + b_left[3 * DH + col];
    float go = acc[4][r] + b_left[4 * DH + col];
    float cv = tanhf(ga) * sigf(gi) + sigf(f1) * s2[DH + col] + sigf(f2) * s1[DH + col];
    float hv = sigf(go) * tanhf(cv);
    if (redF[b]) { s2[col] = hv; s2[DH + col] = cv; }
  }
}

// ---- shift push + pointer update ----
__global__ __launch_bounds__(256) void spinn_update(
    const float* __restrict__ buffers,
    const int* __restrict__ shiftF, const int* __restrict__ redF,
    float* __restrict__ stack, int* __restrict__ sptr, int* __restrict__ blen) {
  int b  = blockIdx.x;
  int sp = sptr[b], bl = blen[b];
  int sh = shiftF[b], rd = redF[b];
  __syncthreads();
  if (sh) {
    const float* bt = buffers + ((long)b * TT + (bl - 1)) * TWO_D;
    float* dst = stack + ((long)b * CAP + sp) * TWO_D;
    for (int i = threadIdx.x; i < TWO_D; i += 256) dst[i] = bt[i];
  }
  if (threadIdx.x == 0) { sptr[b] = sp + sh - rd; blen[b] = bl - sh; }
}

extern "C" void kernel_launch(void* const* d_in, const int* in_sizes, int n_in,
                              void* d_out, int out_size, void* d_ws, size_t ws_size,
                              hipStream_t stream) {
  const float* buffers = (const float*)d_in[0];
  const float* W_left  = (const float*)d_in[1];
  const float* b_left  = (const float*)d_in[2];
  const float* W_right = (const float*)d_in[3];
  const float* W_track = (const float*)d_in[4];
  const float* W_ih    = (const float*)d_in[5];
  const float* W_hh    = (const float*)d_in[6];
  const float* b_ih    = (const float*)d_in[7];
  const float* b_hh    = (const float*)d_in[8];
  const float* W_trans = (const float*)d_in[9];
  const float* b_trans = (const float*)d_in[10];
  int n_steps = in_sizes[11] / BB;     // transitions is [N_TRANS, B]
  float* out = (float*)d_out;

  char* ws = (char*)d_ws;
  size_t off = 0;
  auto carve = [&](size_t bytes) -> void* {
    void* p = ws + off;
    off += (bytes + 255) & ~(size_t)255;
    return p;
  };
  __bf16* Wtrk  = (__bf16*)carve((size_t)KTRK * NTRK * 2);
  __bf16* Wred  = (__bf16*)carve((size_t)KRED * NRED * 2);
  float*  stack = (float*) carve((size_t)BB * CAP * TWO_D * 4);
  __bf16* xh    = (__bf16*)carve((size_t)BB * KTRK * 2);
  __bf16* ybuf  = (__bf16*)carve((size_t)BB * TWO_D * 2);
  __bf16* thb   = (__bf16*)carve((size_t)BB * TRK * 2);
  float*  th    = (float*) carve((size_t)BB * TRK * 4);
  float*  tc    = (float*) carve((size_t)BB * TRK * 4);
  float*  th_n  = (float*) carve((size_t)BB * TRK * 4);
  float*  tc_n  = (float*) carve((size_t)BB * TRK * 4);
  int*    sptr  = (int*)   carve((size_t)BB * 4);
  int*    blen  = (int*)   carve((size_t)BB * 4);
  int*    shF   = (int*)   carve((size_t)BB * 4);
  int*    rdF   = (int*)   carve((size_t)BB * 4);

  long total = (long)KTRK * NTRK + (long)KRED * NRED;
  int prep_blocks = (int)((total + 255) / 256);
  spinn_prep<<<prep_blocks, 256, 0, stream>>>(W_ih, W_hh, W_left, W_right,
                                              W_track, Wtrk, Wred);
  spinn_init<<<BB, 256, 0, stream>>>(buffers, stack, sptr, blen, th, tc);

  for (int s = 0; s < n_steps; ++s) {
    spinn_gather<<<BB, 256, 0, stream>>>(buffers, stack, sptr, blen, th, xh, ybuf);
    spinn_tracker<<<16, 256, 0, stream>>>(xh, Wtrk, b_ih, b_hh, tc, th_n, tc_n);
    spinn_logits<<<1, 128, 0, stream>>>(th_n, tc_n, W_trans, b_trans, sptr, blen,
                                        th, tc, thb, out + (long)s * BB * 4,
                                        shF, rdF);
    spinn_reduce<<<32, 256, 0, stream>>>(ybuf, thb, Wred, b_left, sptr, rdF, stack);
    spinn_update<<<BB, 256, 0, stream>>>(buffers, shF, rdF, stack, sptr, blen);
  }
}
